// MolEncoder_8280696947172
// MI455X (gfx1250) — matrix operand order repaired and checked by reading, not compiled, not run on hardware
//
#include <hip/hip_runtime.h>

// ---------------------------------------------------------------------------
// Types and helpers
// ---------------------------------------------------------------------------
typedef unsigned short u16;
typedef unsigned int   u32;
typedef __attribute__((ext_vector_type(16))) __bf16 v16bf;
typedef __attribute__((ext_vector_type(8)))  float  v8f;
typedef __attribute__((ext_vector_type(4)))  u32    u32x4;

union Frag16 { v16bf v; u32x4 q[2]; };

__device__ __forceinline__ u16 f2bf(float f) {
  union { float f; u32 u; } v; v.f = f;
  u32 r = v.u + 0x7FFFu + ((v.u >> 16) & 1u);   // round-to-nearest-even
  return (u16)(r >> 16);
}
__device__ __forceinline__ float bf2f(u16 h) {
  union { u32 u; float f; } v; v.u = ((u32)h) << 16;
  return v.f;
}

#define HID 256

// ---------------------------------------------------------------------------
// WMMA GEMM: C[M,256] = epilogue( A[M,K](bf16) @ W[256,K].T(bf16) )
// Wave computes a 16x128 strip (8 accumulator tiles). 256-thread workgroup =
// 8 waves = 4 M-tiles x 2 N-halves covering 64 rows x 256 cols.
// Two-stage software pipeline: while WMMAs 0..3 consume the low-half B
// buffers, loads for the high-half buffers (and then the next K-step's A +
// low half) are already in flight. bl/bh have overlapping live ranges, so RA
// is forced to keep two buffer sets -> partial loadcnt waits instead of full
// drains. ~170 VGPRs: no high-VGPR-file MSB toggles, ~5 waves/SIMD.
// Fragment loads are raw b128s; A and B share the same contiguous K order:
//   A (16x32, lane l: m=l&15, g=l>>4): K run      [kb+16g .. +15]
//   B (32x16, lane l: n=l&15, g=l>>4): K run      [kb+16g .. +15]  (weight row n)
//   C (v8f, VGPR r): row = r + 8g, col = n
// A rows must be padded to a multiple of 64 in workspace.
// ---------------------------------------------------------------------------
__global__ __launch_bounds__(256) void gemm_bf16_256(
    const u16* __restrict__ A, const u16* __restrict__ W,
    const float* __restrict__ stat,   // [Mpad,256] per-row additive term, or null
    const float* __restrict__ bias,   // [256] broadcast bias, or null
    float* __restrict__ Cf,           // optional f32 output [M,256]
    u16*   __restrict__ Cb,          // optional bf16 output [Mpad,256]
    int M, int K, int relu, int zero0)
{
  const int lane = threadIdx.x & 31;
  const int wave = threadIdx.x >> 5;
  const int g    = lane >> 4;
  const int nl   = lane & 15;
  const int nh   = (wave & 1) * 128;                       // N half: 0 or 128
  const long m0  = (long)blockIdx.x * 64 + (long)(wave >> 1) * 16;

  v8f acc[8];
#pragma unroll
  for (int i = 0; i < 8; ++i)
#pragma unroll
    for (int r = 0; r < 8; ++r) acc[i][r] = 0.0f;

  // Loop-invariant per-lane base pointers (g-dependent K-offset folded in).
  const u16* arow = A + (m0 + nl) * (long)K + g * 16;
  const u16* wrow[8];
#pragma unroll
  for (int nb = 0; nb < 8; ++nb)
    wrow[nb] = W + (long)(nh + nb * 16 + nl) * K + g * 16;

  // -------- software-pipeline prologue: A + low-half B for kb=0 --------
  Frag16 a, an;
  Frag16 bl[4], bh[4];
  a.q[0] = *(const u32x4*)(arow);
  a.q[1] = *(const u32x4*)(arow + 8);
#pragma unroll
  for (int nb = 0; nb < 4; ++nb) {
    bl[nb].q[0] = *(const u32x4*)(wrow[nb]);
    bl[nb].q[1] = *(const u32x4*)(wrow[nb] + 8);
  }

  for (int kb = 0; kb < K; kb += 32) {
    // stage 1: issue high-half B loads for current kb
#pragma unroll
    for (int nb = 0; nb < 4; ++nb) {
      bh[nb].q[0] = *(const u32x4*)(wrow[nb + 4] + kb);
      bh[nb].q[1] = *(const u32x4*)(wrow[nb + 4] + kb + 8);
    }
    // WMMAs 0..3 consume bl (loaded one stage earlier)
#pragma unroll
    for (int nb = 0; nb < 4; ++nb) {
      acc[nb] = __builtin_amdgcn_wmma_f32_16x16x32_bf16(
          false, a.v, false, bl[nb].v, (short)0, acc[nb], false, false);
    }
    // stage 2: issue next K-step's A + low-half B (clamped on last iter so
    // A never reads past its padded array; values are unused then)
    int kn = kb + 32;
    if (kn >= K) kn = 0;
    an.q[0] = *(const u32x4*)(arow + kn);
    an.q[1] = *(const u32x4*)(arow + kn + 8);
#pragma unroll
    for (int nb = 0; nb < 4; ++nb) {
      bl[nb].q[0] = *(const u32x4*)(wrow[nb] + kn);
      bl[nb].q[1] = *(const u32x4*)(wrow[nb] + kn + 8);
    }
    // WMMAs 4..7 consume bh with current a
#pragma unroll
    for (int nb = 0; nb < 4; ++nb) {
      acc[nb + 4] = __builtin_amdgcn_wmma_f32_16x16x32_bf16(
          false, a.v, false, bh[nb].v, (short)0, acc[nb + 4], false, false);
    }
    a = an;
  }

#pragma unroll
  for (int nb = 0; nb < 8; ++nb) {
    const int n = nh + nb * 16 + nl;
    const float bn = bias ? bias[n] : 0.0f;
#pragma unroll
    for (int r = 0; r < 8; ++r) {
      const long row = m0 + r + 8 * g;
      if (row < M) {
        float v = acc[nb][r] + bn;
        if (stat) v += stat[row * HID + n];
        if (relu && v < 0.0f) v = 0.0f;
        if (zero0 && row == 0) v = 0.0f;
        if (Cf) Cf[row * HID + n] = v;
        if (Cb) Cb[row * HID + n] = f2bf(v);
      }
    }
  }
}

// ---------------------------------------------------------------------------
// Small support kernels (all: grid = rows, block = 256 = hidden dim)
// ---------------------------------------------------------------------------

// fp32 [256,Ks] weight sub-block -> bf16 row-major [256,Kd]
__global__ void k_convw(const float* __restrict__ src, int Ks, int c0,
                        u16* __restrict__ dst, int Kd) {
  long i = (long)blockIdx.x * blockDim.x + threadIdx.x;
  long total = 256L * Kd;
  if (i >= total) return;
  long r = i / Kd, c = i % Kd;
  dst[i] = f2bf(src[r * Ks + c0 + c]);
}

// hstat[e][h] = Wa_w[h, atom(e)] + Wa_w[h, 64+bond(e)] + Wa_b[h]
__global__ void k_hstat(const float* __restrict__ Wa_w, const float* __restrict__ Wa_b,
                        const int* __restrict__ fnode_g, const int* __restrict__ src,
                        const int* __restrict__ bond, float* __restrict__ hstat) {
  long e = blockIdx.x; int h = threadIdx.x;
  int a = fnode_g[src[e]];
  int b = bond[e];
  hstat[e * 256 + h] = Wa_w[h * 324 + a] + Wa_w[h * 324 + 64 + b] + Wa_b[h];
}

// stat_a[v][h] = outA_w[h, atom(v)] + outA_b[h]
__global__ void k_stat_atom(const float* __restrict__ outA_w, const float* __restrict__ outA_b,
                            const int* __restrict__ fnode_g, float* __restrict__ stat) {
  long v = blockIdx.x; int h = threadIdx.x;
  stat[v * 256 + h] = outA_w[h * 1088 + fnode_g[v]] + outA_b[h];
}

// msgs0 = relu(stat), row 0 = 0, bf16 out
__global__ void k_relu_stat(const float* __restrict__ stat, u16* __restrict__ out) {
  long e = blockIdx.x; int h = threadIdx.x;
  float v = stat[e * 256 + h];
  v = v > 0.0f ? v : 0.0f;
  if (e == 0) v = 0.0f;
  out[e * 256 + h] = f2bf(v);
}

// dst[v][col0+h] = sum_k src_bf16[idx[v][k]][h]   (f32 accumulate)
__global__ void k_gather_sum_bf16(const u16* __restrict__ src, const int* __restrict__ idx,
                                  int Kg, int ldidx, u16* __restrict__ dst,
                                  long dstride, long col0) {
  long v = blockIdx.x; int h = threadIdx.x;
  float s = 0.0f;
  for (int k = 0; k < Kg; ++k) {
    long j = idx[v * ldidx + k];
    s += bf2f(src[j * 256 + h]);
  }
  dst[v * dstride + col0 + h] = f2bf(s);
}

// Acat_i[v] = [ bf16(emb[fnode_t[v]]) | bf16(sum_k hatom[dgraph[v][k]]) ]
__global__ void k_embed_tree(const float* __restrict__ emb, const int* __restrict__ fnode_t,
                             const float* __restrict__ hatom, const int* __restrict__ dgraph,
                             u16* __restrict__ Acat) {
  long v = blockIdx.x; int h = threadIdx.x;
  Acat[v * 512 + h] = f2bf(emb[(long)fnode_t[v] * 256 + h]);
  float s = 0.0f;
  for (int k = 0; k < 10; ++k) s += hatom[(long)dgraph[v * 10 + k] * 256 + h];
  Acat[v * 512 + 256 + h] = f2bf(s);
}

// Acat_g[e] = [ hnode_t0[fmess_t_src[e]] | bf16(sum_k hatom[cgraph[e][k]]) ]
__global__ void k_wg_input(const u16* __restrict__ hnode_t0, const int* __restrict__ fsrc,
                           const float* __restrict__ hatom, const int* __restrict__ cgraph,
                           u16* __restrict__ Acat) {
  long e = blockIdx.x; int h = threadIdx.x;
  Acat[e * 512 + h] = hnode_t0[(long)fsrc[e] * 256 + h];
  float s = 0.0f;
  for (int k = 0; k < 8; ++k) s += hatom[(long)cgraph[e * 8 + k] * 256 + h];
  Acat[e * 512 + 256 + h] = f2bf(s);
}

// dst[v*dstride + h] = src[v*256 + h]
__global__ void k_copy_bf16(const u16* __restrict__ src, u16* __restrict__ dst, long dstride) {
  long v = blockIdx.x; int h = threadIdx.x;
  dst[v * dstride + h] = src[v * 256 + h];
}

// embedding[b] = sum_k hnode[revise[b][k]]
__global__ void k_revise(const float* __restrict__ hnode, const int* __restrict__ rn,
                         float* __restrict__ out) {
  long b = blockIdx.x; int h = threadIdx.x;
  float s = 0.0f;
  for (int k = 0; k < 6; ++k) s += hnode[(long)rn[b * 6 + k] * 256 + h];
  out[b * 256 + h] = s;
}

// ---------------------------------------------------------------------------
// Host-side orchestration
// ---------------------------------------------------------------------------
extern "C" void kernel_launch(void* const* d_in, const int* in_sizes, int n_in,
                              void* d_out, int out_size, void* d_ws, size_t ws_size,
                              hipStream_t stream) {
  (void)in_sizes; (void)n_in; (void)out_size; (void)ws_size;

  const float* emb    = (const float*)d_in[0];
  const float* Wa_w   = (const float*)d_in[1];  const float* Wa_b   = (const float*)d_in[2];
  const float* Wi_w   = (const float*)d_in[3];  const float* Wi_b   = (const float*)d_in[4];
  const float* Wg_w   = (const float*)d_in[5];  const float* Wg_b   = (const float*)d_in[6];
  const float* Wt_w   = (const float*)d_in[7];  const float* Wt_b   = (const float*)d_in[8];
  const float* outA_w = (const float*)d_in[9];  const float* outA_b = (const float*)d_in[10];
  const float* outN_w = (const float*)d_in[11]; const float* outN_b = (const float*)d_in[12];
  const int* fnode_g     = (const int*)d_in[13];
  const int* fmess_g_src = (const int*)d_in[14];
  const int* fmess_g_bond= (const int*)d_in[15];
  const int* agraph_g    = (const int*)d_in[16];
  const int* bgraph_g    = (const int*)d_in[17];
  const int* fnode_t     = (const int*)d_in[18];
  const int* fmess_t_src = (const int*)d_in[19];
  const int* agraph_t    = (const int*)d_in[20];
  const int* bgraph_t    = (const int*)d_in[21];
  const int* cgraph_t    = (const int*)d_in[22];
  const int* dgraph_t    = (const int*)d_in[23];
  const int* revise      = (const int*)d_in[24];

  const int NA = 40000, EG = 80000, NT = 20000, ET = 40000, B = 4096;
  const long EGp = 80000;   // 1250*64
  const long NAp = 40064;   // 626*64
  const long ETp = 40064;   // 626*64
  const long NTp = 20096;   // 314*64

  float* out_emb   = (float*)d_out;
  float* out_hnode = out_emb + (size_t)B * 256;
  float* out_hatom = out_hnode + (size_t)NT * 256;

  char* ws = (char*)d_ws;
  size_t off = 0;
  auto take = [&](size_t bytes) -> char* {
    char* p = ws + off;
    off += (bytes + 255) & ~(size_t)255;
    return p;
  };

  // bf16 weight sub-blocks (row-major [256, K])
  u16* w_anei  = (u16*)take(256UL *  256 * 2);   // Wa_w[:,68:324]
  u16* w_outA  = (u16*)take(256UL * 1024 * 2);   // outA_w[:,64:1088]
  u16* w_i     = (u16*)take(256UL *  512 * 2);   // Wi_w
  u16* w_g     = (u16*)take(256UL *  512 * 2);   // Wg_w
  u16* w_tstat = (u16*)take(256UL *  256 * 2);   // Wt_w[:,0:256]
  u16* w_tnei  = (u16*)take(256UL *  256 * 2);   // Wt_w[:,256:512]
  u16* w_outN  = (u16*)take(256UL * 1792 * 2);   // outN_w

  // big fp32 stat buffer, reused sequentially: hstat(EG) -> stat_atom(NA) -> tstat(ET)
  float* statbuf = (float*)take((size_t)EGp * 256 * 4);
  u16* msgs_g  = (u16*)take((size_t)4 * EGp * 256 * 2);
  u16* neibuf  = (u16*)take((size_t)EGp * 256 * 2);      // reused for tree nei
  u16* messnei = (u16*)take((size_t)NAp * 1024 * 2);
  u16* acat_i  = (u16*)take((size_t)NTp * 512 * 2);
  u16* hnodet0 = (u16*)take((size_t)NTp * 256 * 2);
  u16* acat_g  = (u16*)take((size_t)ETp * 512 * 2);
  u16* hmess_t = (u16*)take((size_t)ETp * 256 * 2);
  u16* msgs_t  = (u16*)take((size_t)6 * ETp * 256 * 2);
  u16* acat_n  = (u16*)take((size_t)NTp * 1792 * 2);

  auto cw = [&](const float* src, int Ks, int c0, u16* dst, int Kd) {
    long total = 256L * Kd;
    k_convw<<<(unsigned)((total + 255) / 256), 256, 0, stream>>>(src, Ks, c0, dst, Kd);
  };
  cw(Wa_w,   324,  68, w_anei,   256);
  cw(outA_w, 1088, 64, w_outA,  1024);
  cw(Wi_w,   512,   0, w_i,      512);
  cw(Wg_w,   512,   0, w_g,      512);
  cw(Wt_w,   512,   0, w_tstat,  256);
  cw(Wt_w,   512, 256, w_tnei,   256);
  cw(outN_w, 1792,  0, w_outN,  1792);

  // ---- graph MPNN ----
  k_hstat<<<EG, 256, 0, stream>>>(Wa_w, Wa_b, fnode_g, fmess_g_src, fmess_g_bond, statbuf);
  k_relu_stat<<<EG, 256, 0, stream>>>(statbuf, msgs_g);   // depth 0
  for (int d = 1; d < 4; ++d) {
    k_gather_sum_bf16<<<EG, 256, 0, stream>>>(msgs_g + (size_t)(d - 1) * EGp * 256,
                                              bgraph_g, 6, 6, neibuf, 256, 0);
    gemm_bf16_256<<<(unsigned)(EGp / 64), 256, 0, stream>>>(
        neibuf, w_anei, statbuf, nullptr, nullptr,
        msgs_g + (size_t)d * EGp * 256, EG, 256, 1, 1);
  }
  for (int d = 0; d < 4; ++d)
    k_gather_sum_bf16<<<NA, 256, 0, stream>>>(msgs_g + (size_t)d * EGp * 256,
                                              agraph_g, 6, 6, messnei, 1024, (long)d * 256);
  k_stat_atom<<<NA, 256, 0, stream>>>(outA_w, outA_b, fnode_g, statbuf);
  gemm_bf16_256<<<(unsigned)(NAp / 64), 256, 0, stream>>>(
      messnei, w_outA, statbuf, nullptr, out_hatom, nullptr, NA, 1024, 1, 1);

  // ---- embed tree ----
  k_embed_tree<<<NT, 256, 0, stream>>>(emb, fnode_t, out_hatom, dgraph_t, acat_i);
  gemm_bf16_256<<<(unsigned)(NTp / 64), 256, 0, stream>>>(
      acat_i, w_i, nullptr, Wi_b, nullptr, hnodet0, NT, 512, 0, 0);
  k_wg_input<<<ET, 256, 0, stream>>>(hnodet0, fmess_t_src, out_hatom, cgraph_t, acat_g);
  gemm_bf16_256<<<(unsigned)(ETp / 64), 256, 0, stream>>>(
      acat_g, w_g, nullptr, Wg_b, nullptr, hmess_t, ET, 512, 0, 0);

  // ---- tree MPNN ----
  gemm_bf16_256<<<(unsigned)(ETp / 64), 256, 0, stream>>>(
      hmess_t, w_tstat, nullptr, Wt_b, statbuf, nullptr, ET, 256, 0, 0);   // tstat
  k_relu_stat<<<ET, 256, 0, stream>>>(statbuf, msgs_t);   // depth 0
  for (int d = 1; d < 6; ++d) {
    k_gather_sum_bf16<<<ET, 256, 0, stream>>>(msgs_t + (size_t)(d - 1) * ETp * 256,
                                              bgraph_t, 6, 6, neibuf, 256, 0);
    gemm_bf16_256<<<(unsigned)(ETp / 64), 256, 0, stream>>>(
        neibuf, w_tnei, statbuf, nullptr, nullptr,
        msgs_t + (size_t)d * ETp * 256, ET, 256, 1, 1);
  }

  // ---- output node readout ----
  k_copy_bf16<<<NT, 256, 0, stream>>>(hnodet0, acat_n, 1792);
  for (int d = 0; d < 6; ++d)
    k_gather_sum_bf16<<<NT, 256, 0, stream>>>(msgs_t + (size_t)d * ETp * 256,
                                              agraph_t, 6, 6, acat_n, 1792, 256 + (long)d * 256);
  gemm_bf16_256<<<(unsigned)(NTp / 64), 256, 0, stream>>>(
      acat_n, w_outN, nullptr, outN_b, out_hnode, nullptr, NT, 1792, 1, 1);

  // ---- revise-node embedding ----
  k_revise<<<B, 256, 0, stream>>>(out_hnode, revise, out_emb);
}
